// MixtureOfExperts_16192026706659
// MI455X (gfx1250) — compile-verified, matrix operand order checked by reading, not run on hardware
//
#include <hip/hip_runtime.h>

// ---------------------------------------------------------------------------
// MoE (top-2 of 8 experts), routed bf16-WMMA implementation for gfx1250.
//   N=16384 tokens, D=512, F=2048, E=8, K=2.
// Pipeline: zero counters -> transpose/convert weights to bf16 ->
//           router (top2 + softmax + bucket) ->
//           fused expert FFN (GEMM1+ReLU in LDS, GEMM2, residual+LN) ->
//           deterministic combine.
// MTILE=64 / FC=128: 4 WMMAs per B-fragment -> halves L2 weight traffic
// vs the 32-row tile while keeping VGPRs ~210 and LDS ~85 KB.
// Workspace use ~101 MB.
// ---------------------------------------------------------------------------

#define D_DIM 512
#define F_DIM 2048
#define E_NUM 8
#define MTILE 64          // routed rows per block (4 WMMA M-tiles)
#define FC    128         // F-chunk kept in LDS
#define XPITCH 520        // 512 + 8 bf16 pad (16B-aligned rows, bank spread)
#define HPITCH 136        // 128 + 8 bf16 pad

typedef __bf16 bf16;
typedef __attribute__((ext_vector_type(16))) __bf16 v16bf;
typedef __attribute__((ext_vector_type(8)))  float  v8f;

union Frag16 { v16bf v; unsigned u[8]; uint4 q[2]; };

__device__ __forceinline__ v8f wmma_bf16(v16bf a, v16bf b, v8f c) {
  // emits v_wmma_f32_16x16x32_bf16
  return __builtin_amdgcn_wmma_f32_16x16x32_bf16(false, a, false, b,
                                                 (short)0, c, false, false);
}

// B fragment straight from (pre-transposed) global bf16: lane = output col,
// 16 contiguous-K bf16 split as [kb..kb+7] and [kb+16..kb+23].
__device__ __forceinline__ v16bf load_b_g(const bf16* p) {
  Frag16 f;
  f.q[0] = *(const uint4*)(p);
  f.q[1] = *(const uint4*)(p + 16);
  return f.v;
}

// A fragment from LDS (row-major, padded pitch): lane = row, same K split.
__device__ __forceinline__ v16bf load_a_lds(const unsigned* rowBase, int kbHalf) {
  Frag16 f;
  f.q[0] = *(const uint4*)(rowBase + kbHalf);
  f.q[1] = *(const uint4*)(rowBase + kbHalf + 8);
  return f.v;
}

// ---------------------------------------------------------------------------
__global__ void zero_counts_k(int* counts) {
  if (threadIdx.x < E_NUM) counts[threadIdx.x] = 0;
}

// in: [batch][R][C] fp32 -> out: [batch][C][R] bf16  (tiled 32x32 via LDS)
__global__ __launch_bounds__(256) void transpose_cvt(
    const float* __restrict__ in, bf16* __restrict__ out, int R, int C) {
  __shared__ float tile[32][33];
  const size_t base = (size_t)blockIdx.z * R * C;
  const int x0 = blockIdx.x * 32, y0 = blockIdx.y * 32;
#pragma unroll
  for (int j = 0; j < 4; ++j) {
    int y = y0 + threadIdx.y + j * 8;
    tile[threadIdx.y + j * 8][threadIdx.x] =
        in[base + (size_t)y * C + x0 + threadIdx.x];
  }
  __syncthreads();
#pragma unroll
  for (int j = 0; j < 4; ++j) {
    int yo = x0 + threadIdx.y + j * 8;
    out[base + (size_t)yo * R + y0 + threadIdx.x] =
        (bf16)tile[threadIdx.x][threadIdx.y + j * 8];
  }
}

// One wave32 per token: 8 logits, wave reduce, top-2 + softmax, bucket push.
__global__ __launch_bounds__(256) void moe_router(
    const float* __restrict__ x, const float* __restrict__ rw,
    int* __restrict__ counts, unsigned* __restrict__ bucket,
    float* __restrict__ gates, int N) {
  const int wid = threadIdx.x >> 5;
  const int lane = threadIdx.x & 31;
  const int token = blockIdx.x * 8 + wid;
  if (token >= N) return;
  const float* xr = x + (size_t)token * D_DIM;
  float acc[E_NUM];
#pragma unroll
  for (int e = 0; e < E_NUM; ++e) acc[e] = 0.f;
  for (int d = lane; d < D_DIM; d += 32) {
    const float xv = xr[d];
    const float4* w4 = (const float4*)(rw + d * E_NUM);
    const float4 w0 = w4[0], w1v = w4[1];
    acc[0] += xv * w0.x;  acc[1] += xv * w0.y;
    acc[2] += xv * w0.z;  acc[3] += xv * w0.w;
    acc[4] += xv * w1v.x; acc[5] += xv * w1v.y;
    acc[6] += xv * w1v.z; acc[7] += xv * w1v.w;
  }
#pragma unroll
  for (int e = 0; e < E_NUM; ++e)
#pragma unroll
    for (int m = 16; m >= 1; m >>= 1) acc[e] += __shfl_xor(acc[e], m, 32);
  if (lane == 0) {
    float best = acc[0]; int bi = 0;
#pragma unroll
    for (int e = 1; e < E_NUM; ++e)
      if (acc[e] > best) { best = acc[e]; bi = e; }
    float sec = -3.4e38f; int si = 0;
#pragma unroll
    for (int e = 0; e < E_NUM; ++e)
      if (e != bi && acc[e] > sec) { sec = acc[e]; si = e; }
    const float g0 = 1.f / (1.f + __expf(sec - best));   // softmax over top-2
    gates[2 * token]     = g0;
    gates[2 * token + 1] = 1.f - g0;
    int p0 = atomicAdd(&counts[bi], 1);
    bucket[(size_t)bi * N + p0] = (unsigned)(2 * token);
    int p1 = atomicAdd(&counts[si], 1);
    bucket[(size_t)si * N + p1] = (unsigned)(2 * token + 1);
  }
}

// Fused expert FFN: per block = (expert e, 64 routed rows).
//   GEMM1 (64x512 @ 512xFC) + bias + ReLU -> bf16 H in LDS
//   GEMM2 accumulate (64xFC @ FCx512) over F chunks -> fp32 regs
//   epilogue: +b2 +x residual, cross-wave LayerNorm, scatter to Ybuf[token*2+k].
__global__ __launch_bounds__(256) void moe_expert_ffn(
    const float* __restrict__ x,
    const bf16* __restrict__ W1T,    // [E][F][D] bf16
    const bf16* __restrict__ W2T,    // [E][D][F] bf16
    const float* __restrict__ b1,    // [E][F]
    const float* __restrict__ b2,    // [E][D]
    const float* __restrict__ gamma, // [E][D]
    const float* __restrict__ beta,  // [E][D]
    const int* __restrict__ counts,
    const unsigned* __restrict__ bucket,
    float* __restrict__ Ybuf,        // [N*2][D]
    int N) {
  __shared__ bf16 Xs[MTILE * XPITCH];          // 66.5 KB
  __shared__ bf16 Hs[MTILE * HPITCH];          // 17.4 KB
  __shared__ unsigned rowEntry[MTILE];
  __shared__ float rowSum[MTILE];
  __shared__ float rowSumSq[MTILE];

  const int e = blockIdx.y;
  const int count = counts[e];
  const int rowStart = blockIdx.x * MTILE;
  if (rowStart >= count) return;

  const int tid = threadIdx.x;
  const int lane = tid & 31;
  const int wid = tid >> 5;          // 8 waves
  const int half = lane >> 4;        // K-half select per ISA layout
  const int nl = lane & 15;

  if (tid < MTILE) {
    int r = rowStart + tid;
    rowEntry[tid] = (r < count) ? bucket[(size_t)e * N + r] : 0xFFFFFFFFu;
  }
  __syncthreads();

  // Gather + convert X tile (64 rows x 512) fp32 -> bf16 in LDS.
  {
    const unsigned ent0 = rowEntry[0];
    for (int it = 0; it < 32; ++it) {
      const int idx = tid + it * 256;          // 0..8191 float4 slots
      const int row = idx >> 7;
      const int c4 = idx & 127;
      unsigned ent = rowEntry[row];
      unsigned tok = ((ent == 0xFFFFFFFFu) ? ent0 : ent) >> 1;
      float4 xv = ((const float4*)(x + (size_t)tok * D_DIM))[c4];
      union { bf16 h[4]; uint2 u; } pk;
      pk.h[0] = (bf16)xv.x; pk.h[1] = (bf16)xv.y;
      pk.h[2] = (bf16)xv.z; pk.h[3] = (bf16)xv.w;
      *(uint2*)(&Xs[row * XPITCH + c4 * 4]) = pk.u;
    }
  }
  __syncthreads();

  const unsigned* XsU = (const unsigned*)Xs;
  const unsigned* HsU = (const unsigned*)Hs;
  const v8f vzero = {0.f, 0.f, 0.f, 0.f, 0.f, 0.f, 0.f, 0.f};

  v8f accY[4][4];   // 4 M-tiles x 4 N-tiles (wave owns cols [wid*64, wid*64+64))
#pragma unroll
  for (int mt = 0; mt < 4; ++mt)
#pragma unroll
    for (int j = 0; j < 4; ++j) accY[mt][j] = vzero;

  const bf16* W1e = W1T + (size_t)e * F_DIM * D_DIM;
  const bf16* W2e = W2T + (size_t)e * D_DIM * F_DIM;

  for (int fcIdx = 0; fcIdx < F_DIM / FC; ++fcIdx) {
    const int fc = fcIdx * FC;
    __syncthreads();                 // previous chunk's Hs fully consumed

    // ---- GEMM1: wave owns chunk N-tile {wid}; 4 WMMAs per B fragment ----
    v8f accH[4];
    accH[0] = vzero; accH[1] = vzero; accH[2] = vzero; accH[3] = vzero;
#pragma unroll 4
    for (int kt = 0; kt < D_DIM / 32; ++kt) {
      const int kb = kt * 32 + half * 8;
      v16bf a[4];
#pragma unroll
      for (int mt = 0; mt < 4; ++mt)
        a[mt] = load_a_lds(XsU + (mt * 16 + nl) * (XPITCH / 2), kb >> 1);
      const int f = fc + wid * 16 + nl;
      const v16bf bb = load_b_g(W1e + (size_t)f * D_DIM + kb);
#pragma unroll
      for (int mt = 0; mt < 4; ++mt) accH[mt] = wmma_bf16(a[mt], bb, accH[mt]);
    }
    // bias + ReLU -> bf16 Hs
    {
      const int nc = wid * 16 + nl;
      const float bv = b1[(size_t)e * F_DIM + fc + nc];
#pragma unroll
      for (int mt = 0; mt < 4; ++mt)
#pragma unroll
        for (int r = 0; r < 8; ++r) {
          const int m = mt * 16 + half * 8 + r;
          float h = accH[mt][r] + bv;
          h = h > 0.f ? h : 0.f;
          Hs[m * HPITCH + nc] = (bf16)h;
        }
    }
    __syncthreads();

    // ---- GEMM2 accumulate: wave owns output N-tiles {4*wid..4*wid+3} ----
#pragma unroll
    for (int kt = 0; kt < FC / 32; ++kt) {
      const int kb = kt * 32 + half * 8;
      v16bf a[4];
#pragma unroll
      for (int mt = 0; mt < 4; ++mt)
        a[mt] = load_a_lds(HsU + (mt * 16 + nl) * (HPITCH / 2), kb >> 1);
#pragma unroll
      for (int j = 0; j < 4; ++j) {
        const int n = (4 * wid + j) * 16 + nl;
        const v16bf bb = load_b_g(W2e + (size_t)n * F_DIM + fc + kb);
#pragma unroll
        for (int mt = 0; mt < 4; ++mt)
          accY[mt][j] = wmma_bf16(a[mt], bb, accY[mt][j]);
      }
    }
  }

  // ---- epilogue: residual + LayerNorm + scatter ----
  if (tid < MTILE) { rowSum[tid] = 0.f; rowSumSq[tid] = 0.f; }
  __syncthreads();

#pragma unroll
  for (int mt = 0; mt < 4; ++mt)
#pragma unroll
    for (int j = 0; j < 4; ++j) {
      const int n = (4 * wid + j) * 16 + nl;
      const float b2v = b2[(size_t)e * D_DIM + n];
#pragma unroll
      for (int r = 0; r < 8; ++r) {
        const int m = mt * 16 + half * 8 + r;
        const unsigned ent = rowEntry[m];
        const unsigned tok = (ent == 0xFFFFFFFFu) ? (rowEntry[0] >> 1) : (ent >> 1);
        accY[mt][j][r] += b2v + x[(size_t)tok * D_DIM + n];
      }
    }

  // per-row mean / meansq partials: reduce 16 lanes (same row) then LDS-atomic.
#pragma unroll
  for (int mt = 0; mt < 4; ++mt)
#pragma unroll
    for (int r = 0; r < 8; ++r) {
      float s = 0.f, ss = 0.f;
#pragma unroll
      for (int j = 0; j < 4; ++j) {
        const float v = accY[mt][j][r];
        s += v; ss += v * v;
      }
#pragma unroll
      for (int msk = 8; msk >= 1; msk >>= 1) {
        s  += __shfl_xor(s, msk, 32);
        ss += __shfl_xor(ss, msk, 32);
      }
      if (nl == 0) {
        const int m = mt * 16 + half * 8 + r;
        atomicAdd(&rowSum[m], s);
        atomicAdd(&rowSumSq[m], ss);
      }
    }
  __syncthreads();

#pragma unroll
  for (int mt = 0; mt < 4; ++mt)
#pragma unroll
    for (int j = 0; j < 4; ++j) {
      const int n = (4 * wid + j) * 16 + nl;
      const float gv = gamma[(size_t)e * D_DIM + n];
      const float bv = beta[(size_t)e * D_DIM + n];
#pragma unroll
      for (int r = 0; r < 8; ++r) {
        const int m = mt * 16 + half * 8 + r;
        const unsigned ent = rowEntry[m];
        if (ent != 0xFFFFFFFFu) {
          const float mu  = rowSum[m] * (1.f / D_DIM);
          const float var = rowSumSq[m] * (1.f / D_DIM) - mu * mu;
          const float yv =
              (accY[mt][j][r] - mu) * rsqrtf(var + 1e-6f) * gv + bv;
          Ybuf[(size_t)ent * D_DIM + n] = yv;
        }
      }
    }
}

// out[n] = g0 * Y[n,0] + g1 * Y[n,1]   (fixed order -> deterministic)
__global__ __launch_bounds__(256) void moe_combine(
    const float* __restrict__ Ybuf, const float* __restrict__ gates,
    float* __restrict__ out, int N) {
  const int t = blockIdx.x * 256 + threadIdx.x;   // over N * (D/4)
  if (t >= N * (D_DIM / 4)) return;
  const int n = t >> 7;
  const int c4 = t & 127;
  const float g0 = gates[2 * n], g1 = gates[2 * n + 1];
  const float4 a = ((const float4*)(Ybuf + (size_t)(2 * n) * D_DIM))[c4];
  const float4 b = ((const float4*)(Ybuf + (size_t)(2 * n + 1) * D_DIM))[c4];
  float4 o;
  o.x = g0 * a.x + g1 * b.x;
  o.y = g0 * a.y + g1 * b.y;
  o.z = g0 * a.z + g1 * b.z;
  o.w = g0 * a.w + g1 * b.w;
  ((float4*)out)[(size_t)n * (D_DIM / 4) + c4] = o;
}

// ---------------------------------------------------------------------------
extern "C" void kernel_launch(void* const* d_in, const int* in_sizes, int n_in,
                              void* d_out, int out_size, void* d_ws,
                              size_t ws_size, hipStream_t stream) {
  const float* x        = (const float*)d_in[0];
  const float* router_w = (const float*)d_in[1];
  const float* w1       = (const float*)d_in[2];
  const float* b1       = (const float*)d_in[3];
  const float* w2       = (const float*)d_in[4];
  const float* b2       = (const float*)d_in[5];
  const float* gamma    = (const float*)d_in[6];
  const float* beta     = (const float*)d_in[7];
  float* out = (float*)d_out;
  const int N = in_sizes[0] / D_DIM;   // 16384 tokens

  // workspace carve (all chunks 256B-aligned)
  char* p = (char*)d_ws;
  int* counts = (int*)p;              p += 256;
  unsigned* bucket = (unsigned*)p;    p += (size_t)E_NUM * N * 4;
  float* gates = (float*)p;           p += (size_t)N * 2 * 4;
  bf16* W1T = (bf16*)p;               p += (size_t)E_NUM * F_DIM * D_DIM * 2;
  bf16* W2T = (bf16*)p;               p += (size_t)E_NUM * D_DIM * F_DIM * 2;
  float* Ybuf = (float*)p;            // N*2*D_DIM*4 bytes

  zero_counts_k<<<1, 64, 0, stream>>>(counts);
  transpose_cvt<<<dim3(F_DIM / 32, D_DIM / 32, E_NUM), dim3(32, 8), 0, stream>>>(
      w1, W1T, D_DIM, F_DIM);
  transpose_cvt<<<dim3(D_DIM / 32, F_DIM / 32, E_NUM), dim3(32, 8), 0, stream>>>(
      w2, W2T, F_DIM, D_DIM);
  moe_router<<<(N + 7) / 8, 256, 0, stream>>>(x, router_w, counts, bucket,
                                              gates, N);
  moe_expert_ffn<<<dim3((N + MTILE - 1) / MTILE, E_NUM), 256, 0, stream>>>(
      x, W1T, W2T, b1, b2, gamma, beta, counts, bucket, Ybuf, N);
  moe_combine<<<(N * (D_DIM / 4) + 255) / 256, 256, 0, stream>>>(Ybuf, gates,
                                                                 out, N);
}
